// GATPolicy_70403103916689
// MI455X (gfx1250) — compile-verified, matrix operand order checked by reading, not run on hardware
//
#include <hip/hip_runtime.h>
#include <hip/hip_bf16.h>
#include <hip/hip_fp16.h>

typedef __attribute__((ext_vector_type(16))) _Float16 v16h;
typedef __attribute__((ext_vector_type(8)))  _Float16 half8;
typedef __attribute__((ext_vector_type(8)))  float    v8f;

#define HC   64
#define HEADS 2
#define CH   32
#define ED   5
#define NG   128
#define ACT  8

// ---------------- self loops: deg + sum(edge_attr by dst) ----------------
__global__ void k_deg_loopsum(const int* __restrict__ ei, const float* __restrict__ ea,
                              float* __restrict__ deg, float* __restrict__ lsum, int E) {
    int t = blockIdx.x * blockDim.x + threadIdx.x;
    if (t >= E) return;
    int dst = ei[E + t];
    atomicAdd(&deg[dst], 1.0f);
#pragma unroll
    for (int d = 0; d < ED; ++d)
        atomicAdd(&lsum[dst * ED + d], ea[t * ED + d]);
}

__global__ void k_loopdiv(float* __restrict__ lsum, const float* __restrict__ deg, int Nn) {
    int t = blockIdx.x * blockDim.x + threadIdx.x;
    if (t >= Nn * ED) return;
    int n = t / ED;
    lsum[t] = lsum[t] / fmaxf(deg[n], 1.0f);
}

// ---------------- conversions ----------------
// Wt[n*K + k] = (f16) W[k*HC + n]   (transposed so B-fragment loads are contiguous)
__global__ void k_wt(const float* __restrict__ W, _Float16* __restrict__ Wt, int K) {
    int t = blockIdx.x * blockDim.x + threadIdx.x;
    if (t >= K * HC) return;
    int n = t / K, k = t % K;
    Wt[t] = (_Float16)W[k * HC + n];
}

__global__ void k_tohalf(const float* __restrict__ x, _Float16* __restrict__ y, int n) {
    int t = blockIdx.x * blockDim.x + threadIdx.x;
    if (t < n) y[t] = (_Float16)x[t];
}

// q[d*2+h] = sum_c We[d*HC + h*CH + c] * ae[h*CH + c]
__global__ void k_q(const float* __restrict__ We, const float* __restrict__ ae,
                    float* __restrict__ q) {
    int t = threadIdx.x;
    if (t >= ED * HEADS) return;
    int d = t >> 1, h = t & 1;
    float s = 0.0f;
#pragma unroll
    for (int c = 0; c < CH; ++c) s += We[d * HC + h * CH + c] * ae[h * CH + c];
    q[d * 2 + h] = s;
}

// ---------------- WMMA GEMM: XS[N x 64] = X[N x K] * W[K x 64] ----------------
// block = 128 threads (4 waves); wave w computes the 16x16 tile (blockIdx.x, w).
__global__ void k_gemm(const _Float16* __restrict__ X, const _Float16* __restrict__ Wt,
                       float* __restrict__ XS, int K) {
    int wave  = threadIdx.x >> 5;      // ntile 0..3
    int lane  = threadIdx.x & 31;
    int mtile = blockIdx.x;
    int r  = lane & 15;
    int hh = lane >> 4;
    int row = mtile * 16 + r;          // A row this lane owns
    int col = wave * 16 + r;           // B column this lane owns
    v8f acc = {};
    for (int kb = 0; kb < K; kb += 32) {
        const _Float16* ap = X  + (long)row * K + kb + hh * 8;
        const _Float16* bp = Wt + (long)col * K + kb + hh * 8;
        half8 a0 = *(const half8*)(ap);
        half8 a1 = *(const half8*)(ap + 16);
        half8 b0 = *(const half8*)(bp);
        half8 b1 = *(const half8*)(bp + 16);
        v16h A, B;
#pragma unroll
        for (int j = 0; j < 8; ++j) {
            A[j] = a0[j]; A[j + 8] = a1[j];
            B[j] = b0[j]; B[j + 8] = b1[j];
        }
        acc = __builtin_amdgcn_wmma_f32_16x16x32_f16(
            false, A, false, B, (short)0, acc, false, false);
    }
    float* out = XS + (long)(mtile * 16 + hh * 8) * HC + wave * 16 + r;
#pragma unroll
    for (int i = 0; i < 8; ++i) out[(long)i * HC] = acc[i];
}

// ---------------- per-node attention dots ----------------
__global__ void k_asd(const float* __restrict__ xs, const float* __restrict__ a_s,
                      const float* __restrict__ a_d, float* __restrict__ asrc,
                      float* __restrict__ adst, int Nn) {
    int t = blockIdx.x * blockDim.x + threadIdx.x;
    if (t >= Nn * HEADS) return;
    int n = t >> 1, h = t & 1;
    const float* xp = xs + (long)n * HC + h * CH;
    float ss = 0.0f, sd = 0.0f;
#pragma unroll
    for (int c = 0; c < CH; ++c) {
        float v = xp[c];
        ss += v * a_s[h * CH + c];
        sd += v * a_d[h * CH + c];
    }
    asrc[t] = ss;
    adst[t] = sd;
}

// ---------------- per-edge attr attention dots ----------------
__global__ void k_aedge(const float* __restrict__ ea, const float* __restrict__ lattr,
                        const float* __restrict__ q, float* __restrict__ aedge,
                        int E, int EP) {
    int t = blockIdx.x * blockDim.x + threadIdx.x;
    if (t >= EP) return;
    const float* p = (t < E) ? (ea + (long)t * ED) : (lattr + (long)(t - E) * ED);
#pragma unroll
    for (int h = 0; h < HEADS; ++h) {
        float s = 0.0f;
#pragma unroll
        for (int d = 0; d < ED; ++d) s += p[d] * q[d * 2 + h];
        aedge[(long)t * 2 + h] = s;
    }
}

__device__ __forceinline__ unsigned enc_f(float v) {
    unsigned b = __float_as_uint(v);
    return b ^ ((b >> 31) ? 0xFFFFFFFFu : 0x80000000u);
}

// ---------------- logits + atomic segment max ----------------
__global__ void k_logits_max(const int* __restrict__ ei, const float* __restrict__ asrc,
                             const float* __restrict__ adst, const float* __restrict__ aedge,
                             float* __restrict__ lg, unsigned* __restrict__ mkey,
                             int E, int EP) {
    long t = (long)blockIdx.x * blockDim.x + threadIdx.x;
    if (t >= (long)EP * HEADS) return;
    int e = (int)(t >> 1), h = (int)(t & 1);
    int src = (e < E) ? ei[e] : (e - E);
    int dst = (e < E) ? ei[E + e] : (e - E);
    float v = asrc[src * 2 + h] + adst[dst * 2 + h] + aedge[t];
    v = (v > 0.0f) ? v : 0.2f * v;                  // leaky_relu 0.2
    lg[t] = v;
    atomicMax(&mkey[dst * 2 + h], enc_f(v));
}

__global__ void k_mdecode(unsigned* __restrict__ mkey, int n) {
    int t = blockIdx.x * blockDim.x + threadIdx.x;
    if (t >= n) return;
    unsigned k = mkey[t];
    float m;
    if (k == 0u) m = 0.0f;  // no contributor (cannot happen: self loops)
    else {
        unsigned b = (k >> 31) ? (k ^ 0x80000000u) : ~k;
        m = __uint_as_float(b);
    }
    ((float*)mkey)[t] = m;
}

// ---------------- exp + atomic segment sum ----------------
__global__ void k_exp_den(const int* __restrict__ ei, const float* __restrict__ m,
                          float* __restrict__ lg, float* __restrict__ den, int E, int EP) {
    long t = (long)blockIdx.x * blockDim.x + threadIdx.x;
    if (t >= (long)EP * HEADS) return;
    int e = (int)(t >> 1), h = (int)(t & 1);
    int dst = (e < E) ? ei[E + e] : (e - E);
    float ex = __expf(lg[t] - m[dst * 2 + h]);
    lg[t] = ex;
    atomicAdd(&den[dst * 2 + h], ex);
}

// ---------------- weighted message scatter ----------------
__global__ void k_scatter(const int* __restrict__ ei, const float* __restrict__ xs,
                          const float* __restrict__ lg, const float* __restrict__ den,
                          float* __restrict__ acc, int E, int EP) {
    long t = (long)blockIdx.x * blockDim.x + threadIdx.x;
    if (t >= (long)EP * HC) return;
    int e = (int)(t >> 6), ch = (int)(t & 63), h = ch >> 5;
    int src = (e < E) ? ei[e] : (e - E);
    int dst = (e < E) ? ei[E + e] : (e - E);
    float alpha = lg[(long)e * 2 + h] / (den[dst * 2 + h] + 1e-16f);
    atomicAdd(&acc[(long)dst * HC + ch], xs[(long)src * HC + ch] * alpha);
}

// ---------------- bias + (optional) relu ----------------
__global__ void k_bias_act(const float* __restrict__ acc, const float* __restrict__ b,
                           float* __restrict__ hout, int n, int do_relu) {
    long t = (long)blockIdx.x * blockDim.x + threadIdx.x;
    if (t >= n) return;
    float v = acc[t] + b[t & 63];
    if (do_relu) v = fmaxf(v, 0.0f);
    hout[t] = v;
}

// ---------------- global mean pool ----------------
__global__ void k_pool(const float* __restrict__ h, const int* __restrict__ batch,
                       float* __restrict__ pooled, float* __restrict__ cnt, int Nn) {
    long t = (long)blockIdx.x * blockDim.x + threadIdx.x;
    if (t >= (long)Nn * HC) return;
    int n = (int)(t >> 6), ch = (int)(t & 63);
    int g = batch[n];
    atomicAdd(&pooled[(long)g * HC + ch], h[t]);
    if (ch == 0) atomicAdd(&cnt[g], 1.0f);
}

// ---------------- head: tanh(pooled/cnt @ Wl + bl) ----------------
__global__ void k_head(const float* __restrict__ pooled, const float* __restrict__ cnt,
                       const float* __restrict__ Wl, const float* __restrict__ bl,
                       float* __restrict__ out) {
    int t = blockIdx.x * blockDim.x + threadIdx.x;
    if (t >= NG * ACT) return;
    int g = t >> 3, a = t & 7;
    float inv = 1.0f / fmaxf(cnt[g], 1.0f);
    float s = bl[a];
#pragma unroll
    for (int c = 0; c < HC; ++c) s += pooled[g * HC + c] * inv * Wl[c * ACT + a];
    out[t] = tanhf(s);
}

// =======================================================================
static inline char* carve(char*& p, size_t bytes) {
    char* r = p;
    p += (bytes + 255) & ~(size_t)255;
    return r;
}

extern "C" void kernel_launch(void* const* d_in, const int* in_sizes, int n_in,
                              void* d_out, int out_size, void* d_ws, size_t ws_size,
                              hipStream_t stream) {
    const float* x     = (const float*)d_in[0];
    const int*   ei    = (const int*)d_in[1];
    const float* ea    = (const float*)d_in[2];
    const int*   batch = (const int*)d_in[3];
    const float* W[3]  = {(const float*)d_in[4],  (const float*)d_in[10], (const float*)d_in[16]};
    const float* We[3] = {(const float*)d_in[5],  (const float*)d_in[11], (const float*)d_in[17]};
    const float* As[3] = {(const float*)d_in[6],  (const float*)d_in[12], (const float*)d_in[18]};
    const float* Ad[3] = {(const float*)d_in[7],  (const float*)d_in[13], (const float*)d_in[19]};
    const float* Ae[3] = {(const float*)d_in[8],  (const float*)d_in[14], (const float*)d_in[20]};
    const float* Bi[3] = {(const float*)d_in[9],  (const float*)d_in[15], (const float*)d_in[21]};
    const float* Wl    = (const float*)d_in[22];
    const float* bl    = (const float*)d_in[23];

    const int Nn = in_sizes[0] / 32;       // 50000
    const int E  = in_sizes[1] / 2;        // 800000
    const int EP = E + Nn;                 // with self loops

    char* p = (char*)d_ws;
    float*     deg    = (float*)carve(p, (size_t)Nn * 4);
    float*     lattr  = (float*)carve(p, (size_t)Nn * ED * 4);      // loopsum -> loop_attr
    _Float16*  x16    = (_Float16*)carve(p, (size_t)Nn * HC * 2);
    _Float16*  wt16   = (_Float16*)carve(p, (size_t)HC * HC * 2);
    float*     qbuf   = (float*)carve(p, ED * HEADS * 4);
    float*     xs     = (float*)carve(p, (size_t)Nn * HC * 4);
    float*     hbuf   = (float*)carve(p, (size_t)Nn * HC * 4);
    float*     accum  = (float*)carve(p, (size_t)Nn * HC * 4);
    float*     asrc   = (float*)carve(p, (size_t)Nn * HEADS * 4);
    float*     adst   = (float*)carve(p, (size_t)Nn * HEADS * 4);
    float*     aedge  = (float*)carve(p, (size_t)EP * HEADS * 4);
    float*     lg     = (float*)carve(p, (size_t)EP * HEADS * 4);
    unsigned*  mkey   = (unsigned*)carve(p, (size_t)Nn * HEADS * 4);
    float*     den    = (float*)carve(p, (size_t)Nn * HEADS * 4);
    float*     pooled = (float*)carve(p, (size_t)NG * HC * 4);
    float*     cnt    = (float*)carve(p, (size_t)NG * 4);

    const int B = 256;
    auto g1 = [](long n, int b) { return (unsigned)((n + b - 1) / b); };

    // ---- self loop attrs ----
    hipMemsetAsync(deg,   0, (size_t)Nn * 4, stream);
    hipMemsetAsync(lattr, 0, (size_t)Nn * ED * 4, stream);
    k_deg_loopsum<<<g1(E, B), B, 0, stream>>>(ei, ea, deg, lattr, E);
    k_loopdiv<<<g1((long)Nn * ED, B), B, 0, stream>>>(lattr, deg, Nn);

    for (int l = 0; l < 3; ++l) {
        const int K = (l == 0) ? 32 : HC;
        // projections
        k_wt<<<g1((long)K * HC, B), B, 0, stream>>>(W[l], wt16, K);
        k_q<<<1, 32, 0, stream>>>(We[l], Ae[l], qbuf);
        k_tohalf<<<g1((long)Nn * K, B), B, 0, stream>>>(l == 0 ? x : hbuf, x16, Nn * K);
        k_gemm<<<Nn / 16, 128, 0, stream>>>(x16, wt16, xs, K);   // WMMA path
        // attention terms
        k_asd<<<g1((long)Nn * HEADS, B), B, 0, stream>>>(xs, As[l], Ad[l], asrc, adst, Nn);
        k_aedge<<<g1(EP, B), B, 0, stream>>>(ea, lattr, qbuf, aedge, E, EP);
        // softmax over incoming edges
        hipMemsetAsync(mkey, 0, (size_t)Nn * HEADS * 4, stream);
        hipMemsetAsync(den,  0, (size_t)Nn * HEADS * 4, stream);
        k_logits_max<<<g1((long)EP * HEADS, B), B, 0, stream>>>(ei, asrc, adst, aedge, lg, mkey, E, EP);
        k_mdecode<<<g1((long)Nn * HEADS, B), B, 0, stream>>>(mkey, Nn * HEADS);
        k_exp_den<<<g1((long)EP * HEADS, B), B, 0, stream>>>(ei, (const float*)mkey, lg, den, E, EP);
        // aggregate messages
        hipMemsetAsync(accum, 0, (size_t)Nn * HC * 4, stream);
        k_scatter<<<g1((long)EP * HC, B), B, 0, stream>>>(ei, xs, lg, den, accum, E, EP);
        k_bias_act<<<g1((long)Nn * HC, B), B, 0, stream>>>(accum, Bi[l], hbuf, Nn * HC, l < 2);
    }

    // ---- pool + head ----
    hipMemsetAsync(pooled, 0, (size_t)NG * HC * 4, stream);
    hipMemsetAsync(cnt,    0, (size_t)NG * 4, stream);
    k_pool<<<g1((long)Nn * HC, B), B, 0, stream>>>(hbuf, batch, pooled, cnt, Nn);
    k_head<<<g1(NG * ACT, B), B, 0, stream>>>(pooled, cnt, Wl, bl, (float*)d_out);
}
